// mtransformer_48584670053220
// MI455X (gfx1250) — matrix operand order repaired and checked by reading, not compiled, not run on hardware
//
#include <hip/hip_runtime.h>
#include <hip/hip_bf16.h>
#include <stdint.h>

// ---------------------------------------------------------------------------
// relu-attention + mean-pool, B=8 S=1024 D=1280 H=5 dh=256, gfx1250 (MI455X).
//  - x and W pre-converted to bf16 once (W transposed to [n][k]).
//  - QKV projection + Q@K^T via v_wmma_f32_16x16x32_bf16, f32 accum.
//  - mean-pool folded into column-sums of relu(scores): attn@V becomes a GEMV.
//  - tile staging uses GLOBAL_LOAD_ASYNC_TO_LDS_B128 (ASYNCcnt-tracked).
// ---------------------------------------------------------------------------

typedef __attribute__((ext_vector_type(16))) __bf16 v16bf;
typedef __attribute__((ext_vector_type(8)))  float  v8f;
typedef __attribute__((ext_vector_type(4)))  int    v4i;

union FragAB {
    v16bf        bf;
    unsigned int u[8];
};

#define BM 128
#define BN 128
#define BK 32
#define SA 40   // LDS row stride in ushorts: 80 B -> 16 B aligned rows for b128

// ---- async global->LDS staging (CDNA5) with compile-safe fallback ---------
#if defined(__has_builtin)
#  if __has_builtin(__builtin_amdgcn_global_load_async_to_lds_b128)
#    define ASYNC_LDS 1
#  endif
#endif

typedef __attribute__((address_space(1))) v4i* as1_v4i;
typedef __attribute__((address_space(3))) v4i* as3_v4i;

__device__ __forceinline__ void stage_b128(const void* g, void* l) {
#ifdef ASYNC_LDS
    __builtin_amdgcn_global_load_async_to_lds_b128(
        (as1_v4i)(uintptr_t)g, (as3_v4i)(unsigned)(uintptr_t)l, 0, 0);
#else
    *(uint4*)l = *(const uint4*)g;
#endif
}

__device__ __forceinline__ void stage_wait() {
#ifdef ASYNC_LDS
#  if __has_builtin(__builtin_amdgcn_s_wait_asynccnt)
    __builtin_amdgcn_s_wait_asynccnt(0);
#  else
    asm volatile("s_wait_asynccnt 0" ::: "memory");
#  endif
#endif
}

// ---------------------------------------------------------------------------
__device__ __forceinline__ unsigned short f2bf(float f) {
    unsigned int u = __float_as_uint(f);
    u += 0x7FFFu + ((u >> 16) & 1u);   // round-to-nearest-even
    return (unsigned short)(u >> 16);
}

__device__ __forceinline__ v8f v8f_zero() {
    v8f z = {0.f, 0.f, 0.f, 0.f, 0.f, 0.f, 0.f, 0.f};
    return z;
}

// A-fragment (16x32 bf16): lanes 0-15: V0..3 K=0..7, V4..7 K=16..23; hi lanes +8.
__device__ __forceinline__ int a_koff(int hi, int j) {
    return hi * 8 + ((j < 4) ? 2 * j : 16 + 2 * (j - 4));
}
// B-fragment (32x16 bf16): same K layout as A: lanes 0-15: V0..3 K=0..7, V4..7 K=16..23; hi lanes +8.
__device__ __forceinline__ int b_koff(int hi, int j) {
    return hi * 8 + ((j < 4) ? 2 * j : 16 + 2 * (j - 4));
}

// ---------------------------------------------------------------------------
// Kernel 0a: x fp32 -> bf16 (packed), 4 floats/thread.
// ---------------------------------------------------------------------------
__global__ void __launch_bounds__(256)
cvt_x_kernel(const float4* __restrict__ xin, uint2* __restrict__ xout, int n4)
{
    int i = blockIdx.x * blockDim.x + threadIdx.x;
    if (i >= n4) return;
    float4 f = xin[i];
    unsigned a = (unsigned)f2bf(f.x) | ((unsigned)f2bf(f.y) << 16);
    unsigned b = (unsigned)f2bf(f.z) | ((unsigned)f2bf(f.w) << 16);
    xout[i] = make_uint2(a, b);
}

// ---------------------------------------------------------------------------
// Kernel 0b: W fp32 [k][n] -> bf16 Wt [sel][n][k] (transposed), 32x32 tiles.
// ---------------------------------------------------------------------------
__global__ void __launch_bounds__(256)
cvt_w_kernel(const float* __restrict__ Wq, const float* __restrict__ Wk,
             const float* __restrict__ Wv, unsigned short* __restrict__ Wt)
{
    __shared__ float tile[32][33];
    const float* W = (blockIdx.z == 0) ? Wq : ((blockIdx.z == 1) ? Wk : Wv);
    unsigned short* out = Wt + (size_t)blockIdx.z * 1280 * 1280;
    const int tx = threadIdx.x, ty = threadIdx.y;     // 32 x 8
    const int n0 = blockIdx.x * 32, k0 = blockIdx.y * 32;
#pragma unroll
    for (int j = 0; j < 4; ++j)
        tile[ty + j * 8][tx] = W[(size_t)(k0 + ty + j * 8) * 1280 + (n0 + tx)];
    __syncthreads();
#pragma unroll
    for (int j = 0; j < 4; ++j)
        out[(size_t)(n0 + ty + j * 8) * 1280 + (k0 + tx)] = f2bf(tile[tx][ty + j * 8]);
}

// ---------------------------------------------------------------------------
// Kernel 1: Q/K/V projection, bf16 WMMA. grid = (D/BN, M/BM, 3), 256 thr.
// A = xb [8192][1280] bf16, B = Wt[sel] [n][k] bf16.  Q,K -> bf16, V -> fp32.
// ---------------------------------------------------------------------------
__global__ void __launch_bounds__(256)
qkv_gemm_kernel(const unsigned short* __restrict__ xb,
                const unsigned short* __restrict__ Wt,
                const float* __restrict__ bq, const float* __restrict__ bk,
                const float* __restrict__ bv,
                unsigned short* __restrict__ qb,
                unsigned short* __restrict__ kb,
                float* __restrict__ vf)
{
    const int D = 1280;
    __shared__ unsigned short As[BM * SA];
    __shared__ unsigned short Bs[BN * SA];

    const int t     = threadIdx.x;
    const int lane  = t & 31;
    const int wave  = t >> 5;
    const int wm    = wave & 1;    // 64 rows each
    const int wn    = wave >> 1;   // 32 cols each
    const int row16 = lane & 15;
    const int hi    = lane >> 4;

    const int nblk = blockIdx.x * BN;
    const int mblk = blockIdx.y * BM;
    const int sel  = blockIdx.z;   // 0=Q 1=K 2=V

    const unsigned short* wt = Wt + (size_t)sel * 1280 * 1280;
    const float* bias = (sel == 0) ? bq : ((sel == 1) ? bk : bv);

    v8f acc[4][2];
#pragma unroll
    for (int i = 0; i < 4; ++i)
#pragma unroll
        for (int j = 0; j < 2; ++j) acc[i][j] = v8f_zero();

    for (int kt = 0; kt < 1280 / BK; ++kt) {
        const int k0 = kt * BK;
        // stage A (x) and B (Wt) 128x32 bf16 tiles: 2 x 16B per thread each
#pragma unroll
        for (int i = 0; i < 2; ++i) {
            int lin = i * 256 + t;          // 0..511 16B-chunks
            int r   = lin >> 2;             // row 0..127
            int c4  = lin & 3;              // chunk in row
            stage_b128(&xb[(size_t)(mblk + r) * D + k0 + c4 * 8],
                       &As[r * SA + c4 * 8]);
            stage_b128(&wt[(size_t)(nblk + r) * D + k0 + c4 * 8],
                       &Bs[r * SA + c4 * 8]);
        }
        stage_wait();
        __syncthreads();

        FragAB a[4], b2[2];
#pragma unroll
        for (int tm = 0; tm < 4; ++tm) {
            int rbase = (wm * 64 + tm * 16 + row16) * SA;
#pragma unroll
            for (int j = 0; j < 8; ++j)
                a[tm].u[j] = *(const unsigned int*)&As[rbase + a_koff(hi, j)];
        }
#pragma unroll
        for (int tn = 0; tn < 2; ++tn) {
            int cbase = (wn * 32 + tn * 16 + row16) * SA;
#pragma unroll
            for (int j = 0; j < 8; ++j)
                b2[tn].u[j] = *(const unsigned int*)&Bs[cbase + b_koff(hi, j)];
        }
#pragma unroll
        for (int tm = 0; tm < 4; ++tm)
#pragma unroll
            for (int tn = 0; tn < 2; ++tn)
                acc[tm][tn] = __builtin_amdgcn_wmma_f32_16x16x32_bf16(
                    false, a[tm].bf, false, b2[tn].bf,
                    (short)0, acc[tm][tn], false, false);
        __syncthreads();
    }

    // epilogue: +bias; Q/K as bf16, V as fp32
#pragma unroll
    for (int tn = 0; tn < 2; ++tn) {
        int col = nblk + wn * 32 + tn * 16 + row16;
        float bval = bias[col];
#pragma unroll
        for (int tm = 0; tm < 4; ++tm) {
#pragma unroll
            for (int i = 0; i < 8; ++i) {
                int row = mblk + wm * 64 + tm * 16 + hi * 8 + i;
                float val = acc[tm][tn][i] + bval;
                size_t idx = (size_t)row * D + col;
                if (sel == 0)      qb[idx] = f2bf(val);
                else if (sel == 1) kb[idx] = f2bf(val);
                else               vf[idx] = val;
            }
        }
    }
}

// ---------------------------------------------------------------------------
// Kernel 2: per (b,h) scores = Q_h @ K_h^T / 16, relu, column-sum over q.
// grid = (S/128 ktiles, S/128 qtiles, B*H), block = 256.
// ---------------------------------------------------------------------------
__global__ void __launch_bounds__(256)
attn_colsum_kernel(const unsigned short* __restrict__ qb,
                   const unsigned short* __restrict__ kb,
                   float* __restrict__ colsum)
{
    const int D = 1280, S = 1024, DH = 256;
    __shared__ unsigned short Qs[BM * SA];
    __shared__ unsigned short Ks[BN * SA];
    __shared__ float colpart[128];

    const int t     = threadIdx.x;
    const int lane  = t & 31;
    const int wave  = t >> 5;
    const int wm    = wave & 1;
    const int wn    = wave >> 1;
    const int row16 = lane & 15;
    const int hi    = lane >> 4;

    const int ktile = blockIdx.x;
    const int qtile = blockIdx.y;
    const int bh    = blockIdx.z;
    const int b     = bh / 5;
    const int h     = bh % 5;

    if (t < 128) colpart[t] = 0.f;

    v8f acc[4][2];
#pragma unroll
    for (int i = 0; i < 4; ++i)
#pragma unroll
        for (int j = 0; j < 2; ++j) acc[i][j] = v8f_zero();

    const size_t qrow0 = (size_t)(b * S + qtile * 128);
    const size_t krow0 = (size_t)(b * S + ktile * 128);
    const int    dcol0 = h * DH;

    for (int kt = 0; kt < DH / BK; ++kt) {
        const int d0 = dcol0 + kt * BK;
#pragma unroll
        for (int i = 0; i < 2; ++i) {
            int lin = i * 256 + t;
            int r   = lin >> 2;
            int c4  = lin & 3;
            stage_b128(&qb[(qrow0 + r) * D + d0 + c4 * 8], &Qs[r * SA + c4 * 8]);
            stage_b128(&kb[(krow0 + r) * D + d0 + c4 * 8], &Ks[r * SA + c4 * 8]);
        }
        stage_wait();
        __syncthreads();

        FragAB a[4], b2[2];
#pragma unroll
        for (int tm = 0; tm < 4; ++tm) {
            int rbase = (wm * 64 + tm * 16 + row16) * SA;
#pragma unroll
            for (int j = 0; j < 8; ++j)
                a[tm].u[j] = *(const unsigned int*)&Qs[rbase + a_koff(hi, j)];
        }
#pragma unroll
        for (int tn = 0; tn < 2; ++tn) {
            int cbase = (wn * 32 + tn * 16 + row16) * SA;
#pragma unroll
            for (int j = 0; j < 8; ++j)
                b2[tn].u[j] = *(const unsigned int*)&Ks[cbase + b_koff(hi, j)];
        }
#pragma unroll
        for (int tm = 0; tm < 4; ++tm)
#pragma unroll
            for (int tn = 0; tn < 2; ++tn)
                acc[tm][tn] = __builtin_amdgcn_wmma_f32_16x16x32_bf16(
                    false, a[tm].bf, false, b2[tn].bf,
                    (short)0, acc[tm][tn], false, false);
        __syncthreads();
    }

    // relu + per-column partial sums (column fixed per lane within an n-tile)
    const float scale = 1.0f / 16.0f;   // 1/sqrt(256)
#pragma unroll
    for (int tn = 0; tn < 2; ++tn) {
        int cloc = wn * 32 + tn * 16 + row16;
        float s = 0.f;
#pragma unroll
        for (int tm = 0; tm < 4; ++tm)
#pragma unroll
            for (int i = 0; i < 8; ++i) {
                float v = acc[tm][tn][i] * scale;
                s += (v > 0.f) ? v : 0.f;
            }
        atomicAdd(&colpart[cloc], s);   // ds_add_f32
    }
    __syncthreads();
    if (t < 128)
        atomicAdd(&colsum[(size_t)bh * S + ktile * 128 + t], colpart[t]);
}

// ---------------------------------------------------------------------------
// Kernel 3: out[b, h*256+dv] = (1/S) * sum_k colsum[b,h,k] * V[b,k,h*256+dv]
// ---------------------------------------------------------------------------
__global__ void __launch_bounds__(256)
pool_out_kernel(const float* __restrict__ vf,
                const float* __restrict__ colsum,
                float* __restrict__ out)
{
    const int t = threadIdx.x;
    const int h = blockIdx.x;   // 0..4
    const int b = blockIdx.y;   // 0..7
    const int j = h * 256 + t;

    const float* cs = &colsum[(size_t)(b * 5 + h) * 1024];
    const float* vb = &vf[(size_t)b * 1024 * 1280];

    float acc = 0.f;
    for (int k = 0; k < 1024; ++k)
        acc += cs[k] * vb[(size_t)k * 1280 + j];

    out[b * 1280 + j] = acc * (1.0f / 1024.0f);
}

// ---------------------------------------------------------------------------
extern "C" void kernel_launch(void* const* d_in, const int* in_sizes, int n_in,
                              void* d_out, int out_size, void* d_ws, size_t ws_size,
                              hipStream_t stream)
{
    const float* x  = (const float*)d_in[0];
    const float* Wq = (const float*)d_in[1];
    const float* bq = (const float*)d_in[2];
    const float* Wk = (const float*)d_in[3];
    const float* bk = (const float*)d_in[4];
    const float* Wv = (const float*)d_in[5];
    const float* bv = (const float*)d_in[6];
    float* out = (float*)d_out;

    char* ws = (char*)d_ws;
    // workspace layout (bytes, all 16B aligned):
    //   xb:  bf16 [8192][1280]          @ 0            (20,971,520)
    //   Wt:  bf16 [3][1280][1280]       @ 20,971,520   ( 9,830,400)
    //   qb:  bf16 [8192][1280]          @ 30,801,920   (20,971,520)
    //   kb:  bf16 [8192][1280]          @ 51,773,440   (20,971,520)
    //   vf:  fp32 [8192][1280]          @ 72,744,960   (41,943,040)
    //   colsum: fp32 [40][1024]         @ 114,688,000  (   163,840)
    unsigned short* xb     = (unsigned short*)(ws);
    unsigned short* Wt     = (unsigned short*)(ws + 20971520);
    unsigned short* qb     = (unsigned short*)(ws + 30801920);
    unsigned short* kb     = (unsigned short*)(ws + 51773440);
    float*          vf     = (float*)(ws + 72744960);
    float*          colsum = (float*)(ws + 114688000);

    // 0) one-time conversions: x -> bf16; W -> bf16 transposed [n][k]
    const int n4 = (8192 * 1280) / 4;
    cvt_x_kernel<<<dim3(n4 / 256), 256, 0, stream>>>(
        (const float4*)x, (uint2*)xb, n4);
    cvt_w_kernel<<<dim3(40, 40, 3), dim3(32, 8), 0, stream>>>(Wq, Wk, Wv, Wt);

    // 1) fused QKV projection (z selects Q/K/V)
    qkv_gemm_kernel<<<dim3(1280 / BN, 8192 / BM, 3), 256, 0, stream>>>(
        xb, Wt, bq, bk, bv, qb, kb, vf);

    // 2) relu-score column sums (colsum must start at zero each call)
    (void)hipMemsetAsync(colsum, 0, 40 * 1024 * sizeof(float), stream);
    attn_colsum_kernel<<<dim3(8, 8, 40), 256, 0, stream>>>(qb, kb, colsum);

    // 3) weighted V reduction + mean pool
    pool_out_kernel<<<dim3(5, 8), 256, 0, stream>>>(vf, colsum, out);
}
